// WinPredictorWithPositionalAttention_41523743818165
// MI455X (gfx1250) — matrix operand order repaired and checked by reading, not compiled, not run on hardware
//
#include <hip/hip_runtime.h>
#include <hip/hip_fp16.h>

typedef __attribute__((ext_vector_type(16))) _Float16 v16h;
typedef __attribute__((ext_vector_type(8)))  _Float16 v8h;
typedef __attribute__((ext_vector_type(8)))  float    v8f;

namespace wp {

constexpr int    Bn  = 16384;
constexpr int    Sn  = 5;
constexpr int    Dm  = 256;
constexpr int    NHn = 8;
constexpr int    HDn = 32;
constexpr int    Vn  = 131;
constexpr size_t Mrows = (size_t)Bn * Sn;   // 81920 tokens

// ---------------------------------------------------------------------------
// Fragment loaders (contiguous K order, same for A and B).
// A 16x32 f16: lane L holds row M=L&15, K = 16*(L>>4) + [0,16)
//   -> two b128 loads at +0 and +8 halfs (16 contiguous halfs).
// B 32x16 f16: lane L holds col N=L&15, K = 16*(L>>4) + [0,16)
//   -> two b128 loads at +0 and +8 halfs (16 contiguous halfs).
// ---------------------------------------------------------------------------
__device__ __forceinline__ v16h load_a_frag(const _Float16* p) {
  v8h lo = *(const v8h*)p;
  v8h hi = *(const v8h*)(p + 8);
  return __builtin_shufflevector(lo, hi, 0,1,2,3,4,5,6,7,8,9,10,11,12,13,14,15);
}
__device__ __forceinline__ v16h load_b_frag(const _Float16* p) {
  v8h lo = *(const v8h*)p;
  v8h hi = *(const v8h*)(p + 8);
  return __builtin_shufflevector(lo, hi, 0,1,2,3,4,5,6,7,8,9,10,11,12,13,14,15);
}
__device__ __forceinline__ v8f wmma_f16(v16h a, v16h b, v8f c) {
  return __builtin_amdgcn_wmma_f32_16x16x32_f16(false, a, false, b,
                                                (short)0, c, false, false);
}

// ============================================================================
// WMMA GEMM:  C[M][N] = op( A[M][K](f16) @ W[N][K]^T(f16) + bias[N] )
// Register-tiled: one wave computes a 32x32 C tile (2x2 wmma accumulators),
// so 4 fragment loads feed 4 wmma ops (2 b128 loads per wmma instead of 4).
// 4 waves (N-adjacent 32-wide tiles) per block -> block covers 32x128 of C.
// W is stored (out,in) row-major = per-lane contiguous K -> b128 loads.
// Requires M % 32 == 0, N % 128 == 0, K % 32 == 0.
// ============================================================================
template<bool RELU, bool OUT16, bool OUT32>
__global__ __launch_bounds__(128) void gemm_wmma(
    const _Float16* __restrict__ A, const _Float16* __restrict__ W,
    const float* __restrict__ bias, _Float16* __restrict__ C16,
    float* __restrict__ C32, int N, int K)
{
  const int lane = threadIdx.x;                    // 0..31
  const int tn   = blockIdx.y * 4 + threadIdx.y;   // 32-wide N tile
  const int row0 = blockIdx.x * 32;
  const int col0 = tn * 32;

  const int mr = lane & 15;
  const int kA = (lane >> 4) * 16;
  const int kB = (lane >> 4) * 16;

  const _Float16* ap0 = A + (size_t)(row0 + mr) * K + kA;
  const _Float16* ap1 = A + (size_t)(row0 + 16 + mr) * K + kA;
  const _Float16* bp0 = W + (size_t)(col0 + mr) * K + kB;
  const _Float16* bp1 = W + (size_t)(col0 + 16 + mr) * K + kB;

  v8f acc00 = {0.f,0.f,0.f,0.f,0.f,0.f,0.f,0.f};
  v8f acc01 = {0.f,0.f,0.f,0.f,0.f,0.f,0.f,0.f};
  v8f acc10 = {0.f,0.f,0.f,0.f,0.f,0.f,0.f,0.f};
  v8f acc11 = {0.f,0.f,0.f,0.f,0.f,0.f,0.f,0.f};

  for (int k0 = 0; k0 < K; k0 += 32) {
    v16h a0 = load_a_frag(ap0 + k0);
    v16h a1 = load_a_frag(ap1 + k0);
    v16h b0 = load_b_frag(bp0 + k0);
    v16h b1 = load_b_frag(bp1 + k0);
    acc00 = wmma_f16(a0, b0, acc00);
    acc01 = wmma_f16(a0, b1, acc01);
    acc10 = wmma_f16(a1, b0, acc10);
    acc11 = wmma_f16(a1, b1, acc11);
  }

  // C layout: VGPR r -> lanes 0-15: (M=r, N=lane); lanes 16-31: (M=8+r, N=lane-16)
  const int cB = lane & 15;
  const int rB = (lane >> 4) * 8;
  auto store_tile = [&](const v8f& acc, int ro, int co) {
    const int col = co + cB;
    const float bb = bias[col];
    #pragma unroll
    for (int r = 0; r < 8; ++r) {
      float v = acc[r] + bb;
      if (RELU)  v = fmaxf(v, 0.f);
      if (OUT32) C32[(size_t)(ro + rB + r) * N + col] = v;
      if (OUT16) C16[(size_t)(ro + rB + r) * N + col] = (_Float16)v;
    }
  };
  store_tile(acc00, row0,      col0);
  store_tile(acc01, row0,      col0 + 16);
  store_tile(acc10, row0 + 16, col0);
  store_tile(acc11, row0 + 16, col0 + 16);
}

// ============================================================================
// Row LayerNorm (wave per row, 8 rows per 256-thread block).
// out = relu?( (x [+ y] - mean)/sqrt(var+1e-5) * g + b )
// ============================================================================
template<int PER, bool HAS_ADD, bool RELU, bool OUT32, bool OUT16>
__global__ __launch_bounds__(256) void row_ln(
    const float* __restrict__ X, const float* __restrict__ Yadd,
    const float* __restrict__ g, const float* __restrict__ be,
    float* __restrict__ out32, _Float16* __restrict__ out16, int rows)
{
  const int lane = threadIdx.x & 31;
  const int row  = blockIdx.x * 8 + (threadIdx.x >> 5);
  if (row >= rows) return;
  constexpr int Wd = PER * 32;
  const size_t base = (size_t)row * Wd + lane * PER;

  float v[PER];
  float s = 0.f, ss = 0.f;
  #pragma unroll
  for (int i = 0; i < PER; ++i) {
    float x = X[base + i];
    if (HAS_ADD) x += Yadd[base + i];
    v[i] = x; s += x; ss += x * x;
  }
  #pragma unroll
  for (int m = 16; m > 0; m >>= 1) {
    s  += __shfl_xor(s,  m, 32);
    ss += __shfl_xor(ss, m, 32);
  }
  const float mean = s * (1.f / Wd);
  const float var  = ss * (1.f / Wd) - mean * mean;
  const float inv  = rsqrtf(var + 1e-5f);
  #pragma unroll
  for (int i = 0; i < PER; ++i) {
    const int c = lane * PER + i;
    float y = (v[i] - mean) * inv * g[c] + be[c];
    if (RELU)  y = fmaxf(y, 0.f);
    if (OUT32) out32[base + i] = y;
    if (OUT16) out16[base + i] = (_Float16)y;
  }
}

// ============================================================================
// emb = hero_emb[id] + pos_emb[s]; x = LN(emb) (f32 residual + f16 for GEMM)
// ============================================================================
__global__ __launch_bounds__(256) void embed_ln_kernel(
    const int* __restrict__ ids, const float* __restrict__ hero,
    const float* __restrict__ pos, const float* __restrict__ g,
    const float* __restrict__ be, float* __restrict__ emb32,
    float* __restrict__ x32, _Float16* __restrict__ x16, int rows)
{
  const int lane = threadIdx.x & 31;
  const int row  = blockIdx.x * 8 + (threadIdx.x >> 5);
  if (row >= rows) return;
  const int sp = row % Sn;
  const int id = ids[row];
  const size_t base = (size_t)row * Dm + lane * 8;

  float v[8];
  float s = 0.f, ss = 0.f;
  #pragma unroll
  for (int i = 0; i < 8; ++i) {
    const int c = lane * 8 + i;
    float x = hero[(size_t)id * Dm + c] + pos[(size_t)sp * Dm + c];
    emb32[base + i] = x;
    v[i] = x; s += x; ss += x * x;
  }
  #pragma unroll
  for (int m = 16; m > 0; m >>= 1) {
    s += __shfl_xor(s, m, 32); ss += __shfl_xor(ss, m, 32);
  }
  const float mean = s * (1.f / Dm);
  const float inv  = rsqrtf(ss * (1.f / Dm) - mean * mean + 1e-5f);
  #pragma unroll
  for (int i = 0; i < 8; ++i) {
    const int c = lane * 8 + i;
    const float y = (v[i] - mean) * inv * g[c] + be[c];
    x32[base + i] = y;
    x16[base + i] = (_Float16)y;
  }
}

// ============================================================================
// q = LN(hero_emb[actual] + pos_emb[num_visible])   (f32)
// ============================================================================
__global__ __launch_bounds__(256) void actualq_kernel(
    const int* __restrict__ actual, const int* __restrict__ tids,
    const float* __restrict__ hero, const float* __restrict__ pos,
    const float* __restrict__ g, const float* __restrict__ be,
    float* __restrict__ q32)
{
  const int lane = threadIdx.x & 31;
  const int b = blockIdx.x * 8 + (threadIdx.x >> 5);
  if (b >= Bn) return;
  const int id = actual[b];
  int nv = 0;
  #pragma unroll
  for (int j = 0; j < Sn; ++j) nv += (tids[b * Sn + j] != 0);

  float v[8];
  float s = 0.f, ss = 0.f;
  #pragma unroll
  for (int i = 0; i < 8; ++i) {
    const int c = lane * 8 + i;
    float x = hero[(size_t)id * Dm + c] + pos[(size_t)nv * Dm + c];
    v[i] = x; s += x; ss += x * x;
  }
  #pragma unroll
  for (int m = 16; m > 0; m >>= 1) {
    s += __shfl_xor(s, m, 32); ss += __shfl_xor(ss, m, 32);
  }
  const float mean = s * (1.f / Dm);
  const float inv  = rsqrtf(ss * (1.f / Dm) - mean * mean + 1e-5f);
  #pragma unroll
  for (int i = 0; i < 8; ++i) {
    const int c = lane * 8 + i;
    q32[(size_t)b * Dm + c] = (v[i] - mean) * inv * g[c] + be[c];
  }
}

// ============================================================================
// Self-attention over S=5, hd=32, per-(sample, head, query) thread. Tiny work.
// ============================================================================
__global__ __launch_bounds__(256) void attn_kernel(
    const _Float16* __restrict__ Q, const _Float16* __restrict__ Km,
    const _Float16* __restrict__ Vm, const int* __restrict__ ids,
    _Float16* __restrict__ O)
{
  const int t = blockIdx.x * 256 + threadIdx.x;
  if (t >= Bn * NHn * Sn) return;
  const int qi = t % Sn;
  const int bh = t / Sn;
  const int h  = bh % NHn;
  const int b  = bh / NHn;
  const size_t base = (size_t)b * Sn * Dm + h * HDn;

  int msk[Sn];
  #pragma unroll
  for (int j = 0; j < Sn; ++j) msk[j] = (ids[b * Sn + j] == 0);

  float q[HDn];
  #pragma unroll
  for (int d = 0; d < HDn; ++d) q[d] = (float)Q[base + (size_t)qi * Dm + d];

  float s[Sn];
  float smax = -1e30f;
  #pragma unroll
  for (int j = 0; j < Sn; ++j) {
    float a = 0.f;
    #pragma unroll
    for (int d = 0; d < HDn; ++d) a += q[d] * (float)Km[base + (size_t)j * Dm + d];
    a *= 0.17677669529663687f;                 // 1/sqrt(32)
    if (msk[j]) a = -1e9f;
    s[j] = a; smax = fmaxf(smax, a);
  }
  float den = 0.f;
  #pragma unroll
  for (int j = 0; j < Sn; ++j) { s[j] = __expf(s[j] - smax); den += s[j]; }
  const float invd = 1.f / den;

  float o[HDn];
  #pragma unroll
  for (int d = 0; d < HDn; ++d) o[d] = 0.f;
  #pragma unroll
  for (int j = 0; j < Sn; ++j) {
    const float a = s[j] * invd;
    #pragma unroll
    for (int d = 0; d < HDn; ++d) o[d] += a * (float)Vm[base + (size_t)j * Dm + d];
  }
  #pragma unroll
  for (int d = 0; d < HDn; ++d) O[base + (size_t)qi * Dm + d] = (_Float16)o[d];
}

// ============================================================================
// pooled = sum_s (emb+x)*mask / max(cnt,1)   (f32 + f16)
// ============================================================================
__global__ __launch_bounds__(256) void pool_kernel(
    const int* __restrict__ ids, const float* __restrict__ emb32,
    const float* __restrict__ x32, float* __restrict__ ctx32,
    _Float16* __restrict__ ctx16)
{
  const int lane = threadIdx.x & 31;
  const int b = blockIdx.x * 8 + (threadIdx.x >> 5);
  if (b >= Bn) return;
  float acc[8] = {0.f,0.f,0.f,0.f,0.f,0.f,0.f,0.f};
  int cnt = 0;
  for (int sidx = 0; sidx < Sn; ++sidx) {
    if (ids[b * Sn + sidx] != 0) {
      ++cnt;
      const size_t base = (size_t)(b * Sn + sidx) * Dm + lane * 8;
      #pragma unroll
      for (int i = 0; i < 8; ++i) acc[i] += emb32[base + i] + x32[base + i];
    }
  }
  const float inv = 1.f / (float)max(cnt, 1);   // cnt==0 -> acc==0 -> pooled 0
  #pragma unroll
  for (int i = 0; i < 8; ++i) {
    const size_t o = (size_t)b * Dm + lane * 8 + i;
    const float y = acc[i] * inv;
    ctx32[o] = y; ctx16[o] = (_Float16)y;
  }
}

// ============================================================================
// synergy / counter pair scores (one thread per sample)
// ============================================================================
__global__ void pair_kernel(
    const int* __restrict__ tids, const int* __restrict__ oids,
    const float* __restrict__ syn, const float* __restrict__ cnt,
    float* __restrict__ synT, float* __restrict__ synO, float* __restrict__ ctr)
{
  const int b = blockIdx.x * blockDim.x + threadIdx.x;
  if (b >= Bn) return;
  int tv[Sn], ov[Sn];
  #pragma unroll
  for (int i = 0; i < Sn; ++i) { tv[i] = tids[b*Sn+i]; ov[i] = oids[b*Sn+i]; }
  float st = 0.f, so = 0.f, sc = 0.f;
  #pragma unroll
  for (int i = 0; i < Sn; ++i)
    #pragma unroll
    for (int j = i + 1; j < Sn; ++j) {
      if (tv[i] && tv[j]) st += syn[(size_t)tv[i] * Vn + tv[j]];
      if (ov[i] && ov[j]) so += syn[(size_t)ov[i] * Vn + ov[j]];
    }
  #pragma unroll
  for (int i = 0; i < Sn; ++i)
    #pragma unroll
    for (int j = 0; j < Sn; ++j)
      if (tv[i] && ov[j]) sc += cnt[(size_t)tv[i] * Vn + ov[j]];
  synT[b] = st * 0.1f;
  synO[b] = so * 0.1f;
  ctr[b]  = sc * 0.04f;
}

// ============================================================================
// xcat[B][800] f16 = [team_cross | opp_cross | (q+o1+o2) | syn,osyn,cnt | 0pad]
// ============================================================================
__global__ __launch_bounds__(256) void concat_kernel(
    const float* __restrict__ tc, const float* __restrict__ oc,
    const float* __restrict__ q32, const float* __restrict__ o1,
    const float* __restrict__ o2, const float* __restrict__ synT,
    const float* __restrict__ synO, const float* __restrict__ ctr,
    _Float16* __restrict__ xcat)
{
  const int b = blockIdx.x;
  const int c = threadIdx.x;
  const size_t ob = (size_t)b * 800;
  const size_t ib = (size_t)b * Dm + c;
  xcat[ob + c]       = (_Float16)tc[ib];
  xcat[ob + 256 + c] = (_Float16)oc[ib];
  xcat[ob + 512 + c] = (_Float16)(q32[ib] + o1[ib] + o2[ib]);
  if (c < 32) {
    float v = 0.f;
    if (c == 0) v = synT[b]; else if (c == 1) v = synO[b]; else if (c == 2) v = ctr[b];
    xcat[ob + 768 + c] = (_Float16)v;
  }
}

// ============================================================================
// out[b] = x[b,:] . w_out + b_out    (wave per sample, K=512)
// ============================================================================
__global__ __launch_bounds__(256) void outdot_kernel(
    const _Float16* __restrict__ x, const float* __restrict__ w,
    const float* __restrict__ b0, float* __restrict__ out)
{
  const int lane = threadIdx.x & 31;
  const int b = blockIdx.x * 8 + (threadIdx.x >> 5);
  if (b >= Bn) return;
  float s = 0.f;
  #pragma unroll
  for (int i = 0; i < 16; ++i) {
    const int c = lane * 16 + i;
    s += (float)x[(size_t)b * 512 + c] * w[c];
  }
  #pragma unroll
  for (int m = 16; m > 0; m >>= 1) s += __shfl_xor(s, m, 32);
  if (lane == 0) out[b] = s + b0[0];
}

// ============================================================================
// Weight conversion f32 -> f16 (optionally with K padding)
// ============================================================================
__global__ void cvt16_kernel(const float* __restrict__ src,
                             _Float16* __restrict__ dst, int n)
{
  const int i = blockIdx.x * 256 + threadIdx.x;
  if (i < n) dst[i] = (_Float16)src[i];
}

__global__ void cvt16_pad_kernel(const float* __restrict__ src,
                                 _Float16* __restrict__ dst,
                                 int N, int Ks, int Kd)
{
  const int i = blockIdx.x * 256 + threadIdx.x;
  if (i >= N * Kd) return;
  const int n = i / Kd, k = i % Kd;
  dst[i] = (k < Ks) ? (_Float16)src[(size_t)n * Ks + k] : (_Float16)0.f;
}

} // namespace wp

// =============================================================================
// Orchestration.  d_in flattening (insertion order of setup_inputs):
//   0 team_ids  1 opp_ids  2 actual_ids
//   3 hero_emb  4 pos_emb  5 synergy  6 counter
//   7 ln1_g 8 ln1_b 9 ln_ct_g 10 ln_ct_b 11 ln_co_g 12 ln_co_b
//   13..20 ca_team{wq,bq,wk,bk,wv,bv,wo,bo}  21..28 ca_opp  29..36 a2t  37..44 a2o
//   45..60 enc_layer0 {attn(8), ln1_g, ln1_b, w1, b1, w2, b2, ln2_g, ln2_b}
//   61..76 enc_layer1
//   77..80 net0{w,b,g,beta}  81..84 net1  85 w_out  86 b_out
// =============================================================================
extern "C" void kernel_launch(void* const* d_in, const int* in_sizes, int n_in,
                              void* d_out, int out_size, void* d_ws, size_t ws_size,
                              hipStream_t stream)
{
  using namespace wp;
  (void)in_sizes; (void)n_in; (void)out_size; (void)ws_size;

  const int* tids   = (const int*)d_in[0];
  const int* oids   = (const int*)d_in[1];
  const int* actual = (const int*)d_in[2];
  auto F = [&](int i) -> const float* { return (const float*)d_in[i]; };

  const int HERO = 3, POS = 4, SYN = 5, CNT = 6;
  const int LN1G = 7, LN1B = 8, LNCTG = 9, LNCTB = 10, LNCOG = 11, LNCOB = 12;
  const int CAT = 13, CAO = 21, A2T = 29, A2O = 37;   // +4 wv, +5 bv, +6 wo, +7 bo
  const int encB[2] = {45, 61};                        // +1 bq,+3 bk,+5 bv,+7 bo,...
  const int NET0 = 77, NET1 = 81, WOUT = 85, BOUT = 86;

  char* base = (char*)d_ws;
  size_t off = 0;
  auto take = [&](size_t bytes) -> char* {
    char* p = base + off;
    off = (off + bytes + 255) & ~(size_t)255;
    return p;
  };

  // ---- persistent f16 weights ----
  _Float16 *wq16[2], *wk16[2], *wv16[2], *wo16[2], *w1_16[2], *w2_16[2];
  for (int l = 0; l < 2; ++l) {
    wq16[l] = (_Float16*)take(65536 * 2);
    wk16[l] = (_Float16*)take(65536 * 2);
    wv16[l] = (_Float16*)take(65536 * 2);
    wo16[l] = (_Float16*)take(65536 * 2);
    w1_16[l] = (_Float16*)take(262144 * 2);
    w2_16[l] = (_Float16*)take(262144 * 2);
  }
  _Float16* catv16 = (_Float16*)take(65536 * 2);
  _Float16* cato16 = (_Float16*)take(65536 * 2);
  _Float16* cotv16 = (_Float16*)take(65536 * 2);
  _Float16* coto16 = (_Float16*)take(65536 * 2);
  _Float16* a2tv16 = (_Float16*)take(65536 * 2);
  _Float16* a2to16 = (_Float16*)take(65536 * 2);
  _Float16* a2ov16 = (_Float16*)take(65536 * 2);
  _Float16* a2oo16 = (_Float16*)take(65536 * 2);
  _Float16* net0w16 = (_Float16*)take((size_t)1024 * 800 * 2);
  _Float16* net1w16 = (_Float16*)take((size_t)512 * 1024 * 2);

  // ---- persistent pooled contexts (must survive both encoder passes) ----
  float*    tctx32 = (float*)take((size_t)Bn * Dm * 4);
  _Float16* tctx16 = (_Float16*)take((size_t)Bn * Dm * 2);
  float*    octx32 = (float*)take((size_t)Bn * Dm * 4);
  _Float16* octx16 = (_Float16*)take((size_t)Bn * Dm * 2);
  float* synT = (float*)take(Bn * 4);
  float* synO = (float*)take(Bn * 4);
  float* ctrS = (float*)take(Bn * 4);

  // ---- big encoder scratch (reused by both sides, then overlaid post-phase) ----
  const size_t scratch0 = off;
  float*    emb32 = (float*)take(Mrows * Dm * 4);
  float*    x32   = (float*)take(Mrows * Dm * 4);
  _Float16* x16   = (_Float16*)take(Mrows * Dm * 2);
  float*    y32   = (float*)take(Mrows * Dm * 4);
  _Float16* big   = (_Float16*)take(Mrows * 1024 * 2);
  _Float16* q16  = big;
  _Float16* k16  = big + Mrows * Dm;
  _Float16* v16  = big + 2 * Mrows * Dm;
  _Float16* o16  = big + 3 * Mrows * Dm;
  _Float16* h116 = big;                       // FFN hidden reuses q/k/v/o space

  // ---- convert weights to f16 ----
  auto cvt = [&](int idx, _Float16* dst, int n) {
    cvt16_kernel<<<dim3((n + 255) / 256), dim3(256), 0, stream>>>(F(idx), dst, n);
  };
  for (int l = 0; l < 2; ++l) {
    cvt(encB[l] + 0,  wq16[l], 65536);
    cvt(encB[l] + 2,  wk16[l], 65536);
    cvt(encB[l] + 4,  wv16[l], 65536);
    cvt(encB[l] + 6,  wo16[l], 65536);
    cvt(encB[l] + 10, w1_16[l], 262144);
    cvt(encB[l] + 12, w2_16[l], 262144);
  }
  cvt(CAT + 4, catv16, 65536);  cvt(CAT + 6, cato16, 65536);
  cvt(CAO + 4, cotv16, 65536);  cvt(CAO + 6, coto16, 65536);
  cvt(A2T + 4, a2tv16, 65536);  cvt(A2T + 6, a2to16, 65536);
  cvt(A2O + 4, a2ov16, 65536);  cvt(A2O + 6, a2oo16, 65536);
  cvt16_pad_kernel<<<dim3((1024 * 800 + 255) / 256), dim3(256), 0, stream>>>(
      F(NET0 + 0), net0w16, 1024, 771, 800);
  cvt(NET1 + 0, net1w16, 512 * 1024);

  // ---- encoder pass (shared scratch) ----
  const dim3 blk(32, 4);
  auto run_encoder = [&](const int* ids, float* ctx32, _Float16* ctx16) {
    const int rows = (int)Mrows;
    const dim3 rgrid((rows + 7) / 8);
    embed_ln_kernel<<<rgrid, 256, 0, stream>>>(ids, F(HERO), F(POS), F(LN1G),
                                               F(LN1B), emb32, x32, x16, rows);
    const dim3 g256((int)(Mrows / 32), 256 / 128);    // 32x128 C per block
    const dim3 g1024((int)(Mrows / 32), 1024 / 128);
    for (int l = 0; l < 2; ++l) {
      const int pb = encB[l];
      gemm_wmma<false, true, false><<<g256, blk, 0, stream>>>(
          x16, wq16[l], F(pb + 1), q16, nullptr, 256, 256);
      gemm_wmma<false, true, false><<<g256, blk, 0, stream>>>(
          x16, wk16[l], F(pb + 3), k16, nullptr, 256, 256);
      gemm_wmma<false, true, false><<<g256, blk, 0, stream>>>(
          x16, wv16[l], F(pb + 5), v16, nullptr, 256, 256);
      attn_kernel<<<dim3((Bn * NHn * Sn + 255) / 256), 256, 0, stream>>>(
          q16, k16, v16, ids, o16);
      gemm_wmma<false, false, true><<<g256, blk, 0, stream>>>(
          o16, wo16[l], F(pb + 7), nullptr, y32, 256, 256);
      row_ln<8, true, false, true, true><<<rgrid, 256, 0, stream>>>(
          x32, y32, F(pb + 8), F(pb + 9), x32, x16, rows);
      gemm_wmma<true, true, false><<<g1024, blk, 0, stream>>>(
          x16, w1_16[l], F(pb + 11), h116, nullptr, 1024, 256);
      gemm_wmma<false, false, true><<<g256, blk, 0, stream>>>(
          h116, w2_16[l], F(pb + 13), nullptr, y32, 256, 1024);
      row_ln<8, true, false, true, true><<<rgrid, 256, 0, stream>>>(
          x32, y32, F(pb + 14), F(pb + 15), x32, x16, rows);
    }
    pool_kernel<<<dim3((Bn + 7) / 8), 256, 0, stream>>>(ids, emb32, x32, ctx32, ctx16);
  };
  run_encoder(tids, tctx32, tctx16);
  run_encoder(oids, octx32, octx16);

  // ---- post phase: overlay buffers on the (now dead) encoder scratch ----
  size_t poff = scratch0;
  auto ptake = [&](size_t bytes) -> char* {
    char* p = base + poff;
    poff = (poff + bytes + 255) & ~(size_t)255;
    return p;
  };
  float*    tcr32 = (float*)ptake((size_t)Bn * Dm * 4);
  _Float16* tcr16 = (_Float16*)ptake((size_t)Bn * Dm * 2);
  float*    ocr32 = (float*)ptake((size_t)Bn * Dm * 4);
  _Float16* ocr16 = (_Float16*)ptake((size_t)Bn * Dm * 2);
  float*    q32   = (float*)ptake((size_t)Bn * Dm * 4);
  float*    o1    = (float*)ptake((size_t)Bn * Dm * 4);
  float*    o2    = (float*)ptake((size_t)Bn * Dm * 4);
  _Float16* u16   = (_Float16*)ptake((size_t)Bn * Dm * 2);
  _Float16* xcat  = (_Float16*)ptake((size_t)Bn * 800 * 2);
  float*    n132  = (float*)ptake((size_t)Bn * 1024 * 4);
  _Float16* n116  = (_Float16*)ptake((size_t)Bn * 1024 * 2);
  float*    n232  = (float*)ptake((size_t)Bn * 512 * 4);
  _Float16* n216  = (_Float16*)ptake((size_t)Bn * 512 * 2);

  pair_kernel<<<dim3((Bn + 255) / 256), 256, 0, stream>>>(
      tids, oids, F(SYN), F(CNT), synT, synO, ctrS);

  const dim3 gB(Bn / 32, 256 / 128);
  const dim3 bgrid((Bn + 7) / 8);
  // L_k == 1 => softmax == 1 => mha(q,k,v) = (v@wv^T+bv)@wo^T+bo  (q-independent)
  // ca_team: t2o = vo(opp_ctx); team_cross = LN(team_ctx + t2o)
  gemm_wmma<false, true, false><<<gB, blk, 0, stream>>>(
      octx16, catv16, F(CAT + 5), u16, nullptr, 256, 256);
  gemm_wmma<false, false, true><<<gB, blk, 0, stream>>>(
      u16, cato16, F(CAT + 7), nullptr, o1, 256, 256);
  row_ln<8, true, false, true, true><<<bgrid, 256, 0, stream>>>(
      tctx32, o1, F(LNCTG), F(LNCTB), tcr32, tcr16, Bn);
  // ca_opp: o2t = vo(team_ctx); opp_cross = LN(opp_ctx + o2t)
  gemm_wmma<false, true, false><<<gB, blk, 0, stream>>>(
      tctx16, cotv16, F(CAO + 5), u16, nullptr, 256, 256);
  gemm_wmma<false, false, true><<<gB, blk, 0, stream>>>(
      u16, coto16, F(CAO + 7), nullptr, o2, 256, 256);
  row_ln<8, true, false, true, true><<<bgrid, 256, 0, stream>>>(
      octx32, o2, F(LNCOG), F(LNCOB), ocr32, ocr16, Bn);
  // ac = q + vo_a2t(team_cross) + vo_a2o(opp_cross)
  actualq_kernel<<<bgrid, 256, 0, stream>>>(actual, tids, F(HERO), F(POS),
                                            F(LN1G), F(LN1B), q32);
  gemm_wmma<false, true, false><<<gB, blk, 0, stream>>>(
      tcr16, a2tv16, F(A2T + 5), u16, nullptr, 256, 256);
  gemm_wmma<false, false, true><<<gB, blk, 0, stream>>>(
      u16, a2to16, F(A2T + 7), nullptr, o1, 256, 256);
  gemm_wmma<false, true, false><<<gB, blk, 0, stream>>>(
      ocr16, a2ov16, F(A2O + 5), u16, nullptr, 256, 256);
  gemm_wmma<false, false, true><<<gB, blk, 0, stream>>>(
      u16, a2oo16, F(A2O + 7), nullptr, o2, 256, 256);

  concat_kernel<<<dim3(Bn), 256, 0, stream>>>(tcr32, ocr32, q32, o1, o2,
                                              synT, synO, ctrS, xcat);
  // MLP head
  gemm_wmma<false, false, true><<<dim3(Bn / 32, 1024 / 128), blk, 0, stream>>>(
      xcat, net0w16, F(NET0 + 1), nullptr, n132, 1024, 800);
  row_ln<32, false, true, false, true><<<bgrid, 256, 0, stream>>>(
      n132, nullptr, F(NET0 + 2), F(NET0 + 3), nullptr, n116, Bn);
  gemm_wmma<false, false, true><<<dim3(Bn / 32, 512 / 128), blk, 0, stream>>>(
      n116, net1w16, F(NET1 + 1), nullptr, n232, 512, 1024);
  row_ln<16, false, true, false, true><<<bgrid, 256, 0, stream>>>(
      n232, nullptr, F(NET1 + 2), F(NET1 + 3), nullptr, n216, Bn);
  outdot_kernel<<<bgrid, 256, 0, stream>>>(n216, F(WOUT), F(BOUT), (float*)d_out);
}
